// UPET_2826088481160
// MI455X (gfx1250) — matrix operand order repaired and checked by reading, not compiled, not run on hardware
//
#include <hip/hip_runtime.h>
#include <hip/hip_fp16.h>

// ---------------- problem constants ----------------
#define NA    4096          // atoms
#define NNBR  32            // neighbors per atom
#define NP    (NA*NNBR)     // pairs = 131072
#define TTOK  33            // tokens per atom (1 center + 32 edges)
#define MT    (NA*TTOK)     // token rows = 135168 (multiple of 64)
#define NHEADS 8
#define CDIV(a,b) (((a)+(b)-1)/(b))

typedef __attribute__((ext_vector_type(16))) _Float16 v16h;
typedef __attribute__((ext_vector_type(8)))  _Float16 v8h;
typedef __attribute__((ext_vector_type(8)))  float    v8f;

union V16H { v16h v; v8h h[2]; };

__device__ __forceinline__ float fsigmoid(float x) { return 1.f / (1.f + __expf(-x)); }
__device__ __forceinline__ float fsilu(float x)    { return x * fsigmoid(x); }

// =====================================================================
// WMMA GEMM:  C[M,N] = act(A[M,K] @ W + b)
//   A  : f16 row-major [M,K]
//   Wt : f16, weights pre-transposed to [N,K] so B fragments are contiguous
// Fragment layouts per CDNA5 ISA 7.12.2 (wave32):
//   A 16x32 f16 : lane<16 -> M=lane, K in {0..7,16..23}; lane>=16 -> K in {8..15,24..31}
//   B 32x16 f16 : lane<16 -> N=lane, K in {0..7,16..23}; lane>=16 -> K in {8..15,24..31}
//   C 16x16 f32 : vgpr v, lane -> M = v + (lane<16?0:8), N = lane&15
// =====================================================================

// ---- 64x64 tile per wave: 4 A-frags x 4 B-frags -> 16 wmma per K-step.
//      A re-read N/64 times, B re-read M/64 times (4x less traffic than 16x16),
//      and a 1:1 b128-load : wmma ratio keeps the matrix pipe saturated.
template<int ACT>   // 0 = none, 1 = silu
__global__ __launch_bounds__(128) void k_gemm64(
    const _Float16* __restrict__ A, const _Float16* __restrict__ Wt,
    const float* __restrict__ bias,
    float* __restrict__ Cf, _Float16* __restrict__ Ch,
    int M, int N, int K)
{
  const int lane = threadIdx.x & 31;
  const int wid  = threadIdx.x >> 5;
  const int ntiles = N >> 6;
  const int wg = blockIdx.x * 4 + wid;
  if (wg >= (M >> 6) * ntiles) return;            // wave-uniform guard (EXEC all-1)
  const int tm = wg / ntiles, tn = wg % ntiles;
  const int hi = lane >> 4, l15 = lane & 15;

  const _Float16* pa[4];
  const _Float16* pb[4];
  #pragma unroll
  for (int i = 0; i < 4; ++i)
    pa[i] = A  + (size_t)(tm * 64 + i * 16 + l15) * K + (hi ? 8  : 0);
  #pragma unroll
  for (int j = 0; j < 4; ++j)
    pb[j] = Wt + (size_t)(tn * 64 + j * 16 + l15) * K + (hi ? 8  : 0);

  const v8f z = {0.f,0.f,0.f,0.f,0.f,0.f,0.f,0.f};
  v8f acc[16];
  #pragma unroll
  for (int t = 0; t < 16; ++t) acc[t] = z;

  for (int k0 = 0; k0 < K; k0 += 32) {
    V16H a[4], b[4];
    #pragma unroll
    for (int i = 0; i < 4; ++i) {
      __builtin_prefetch(pa[i] + k0 + 128, 0, 1);
      a[i].h[0] = *(const v8h*)(pa[i] + k0);        // K = off+0..7
      a[i].h[1] = *(const v8h*)(pa[i] + k0 + 16);   // K = off+16..23
    }
    #pragma unroll
    for (int j = 0; j < 4; ++j) {
      __builtin_prefetch(pb[j] + k0 + 128, 0, 1);
      b[j].h[0] = *(const v8h*)(pb[j] + k0);        // K = off+0..7
      b[j].h[1] = *(const v8h*)(pb[j] + k0 + 16);   // K = off+16..23
    }
    #pragma unroll
    for (int i = 0; i < 4; ++i)
      #pragma unroll
      for (int j = 0; j < 4; ++j)
        acc[i * 4 + j] = __builtin_amdgcn_wmma_f32_16x16x32_f16(
            false, a[i].v, false, b[j].v, (short)0, acc[i * 4 + j], false, false);
  }

  float bv[4];
  #pragma unroll
  for (int j = 0; j < 4; ++j) bv[j] = bias ? bias[tn * 64 + j * 16 + l15] : 0.f;

  #pragma unroll
  for (int i = 0; i < 4; ++i) {
    #pragma unroll
    for (int v = 0; v < 8; ++v) {
      const int row = tm * 64 + i * 16 + v + (hi ? 8 : 0);
      #pragma unroll
      for (int j = 0; j < 4; ++j) {
        float x = acc[i * 4 + j][v] + bv[j];
        if (ACT == 1) x = fsilu(x);
        const size_t o = (size_t)row * N + tn * 64 + j * 16 + l15;
        if (Cf) Cf[o] = x;
        if (Ch) Ch[o] = (_Float16)x;
      }
    }
  }
}

// ---- 16x16 fallback for shapes not divisible by 64 ----
template<int ACT>
__global__ __launch_bounds__(128) void k_gemm16(
    const _Float16* __restrict__ A, const _Float16* __restrict__ Wt,
    const float* __restrict__ bias,
    float* __restrict__ Cf, _Float16* __restrict__ Ch,
    int M, int N, int K)
{
  const int lane = threadIdx.x & 31;
  const int wid  = threadIdx.x >> 5;
  const int ntiles = N >> 4;
  const int wg = blockIdx.x * 4 + wid;
  if (wg >= (M >> 4) * ntiles) return;
  const int tm = wg / ntiles, tn = wg % ntiles;
  const int hi = lane >> 4, l15 = lane & 15;

  const _Float16* pa = A  + (size_t)(tm * 16 + l15) * K + (hi ? 8  : 0);
  const _Float16* pb = Wt + (size_t)(tn * 16 + l15) * K + (hi ? 8  : 0);

  v8f acc = {0.f,0.f,0.f,0.f,0.f,0.f,0.f,0.f};
  #pragma unroll 4
  for (int k0 = 0; k0 < K; k0 += 32) {
    __builtin_prefetch(pa + k0 + 128, 0, 1);
    __builtin_prefetch(pb + k0 + 128, 0, 1);
    V16H a, b;
    a.h[0] = *(const v8h*)(pa + k0);
    a.h[1] = *(const v8h*)(pa + k0 + 16);
    b.h[0] = *(const v8h*)(pb + k0);
    b.h[1] = *(const v8h*)(pb + k0 + 16);
    acc = __builtin_amdgcn_wmma_f32_16x16x32_f16(
            false, a.v, false, b.v, (short)0, acc, false, false);
  }

  const int col = tn * 16 + l15;
  const float bv = bias ? bias[col] : 0.f;
  #pragma unroll
  for (int v = 0; v < 8; ++v) {
    const int row = tm * 16 + v + (hi ? 8 : 0);
    float x = acc[v] + bv;
    if (ACT == 1) x = fsilu(x);
    const size_t o = (size_t)row * N + col;
    if (Cf) Cf[o] = x;
    if (Ch) Ch[o] = (_Float16)x;
  }
}

// weight convert f32[K,N] -> f16[N,K] (transposed for contiguous B fragments)
__global__ void k_wt_conv(const float* __restrict__ W, _Float16* __restrict__ Wt, int K, int N) {
  int i = blockIdx.x * blockDim.x + threadIdx.x;
  if (i >= K * N) return;
  int k = i / N, n = i - k * N;
  Wt[(size_t)n * K + k] = (_Float16)W[i];
}

// ---------------- prep kernels ----------------
__global__ void k_pair_prep(const float* __restrict__ R, const unsigned char* __restrict__ pm,
                            float* __restrict__ geom, float* __restrict__ cut) {
  int p = blockIdx.x * blockDim.x + threadIdx.x;
  if (p >= NP) return;
  float x = R[p*3+0], y = R[p*3+1], z = R[p*3+2];
  float r = sqrtf(x*x + y*y + z*z + 1e-15f);
  geom[p*4+0] = x; geom[p*4+1] = y; geom[p*4+2] = z; geom[p*4+3] = r;
  float t  = (r - 7.0f) * 2.0f;                       // (r-(cut-w))/w, w=0.5
  float xs = fminf(fmaxf(t, 1e-6f), 1.f - 1e-6f);
  float bump = 0.5f * (1.f + tanhf(1.f / __tanf(3.14159265358979f * xs)));
  float c = (t <= 0.f) ? 1.f : ((t >= 1.f) ? 0.f : bump);
  cut[p] = c * (pm[p] ? 1.f : 0.f);
}

__global__ void k_logcut(const float* __restrict__ cut, float* __restrict__ lc) {
  int i = blockIdx.x * blockDim.x + threadIdx.x;
  if (i >= NA * TTOK) return;
  int t = i % TTOK, a = i / TTOK;
  float c = (t == 0) ? 1.f : cut[a * NNBR + t - 1];
  lc[i] = __logf(fmaxf(c, 1e-15f));
}

// table[species[nbr[p]]] -> f32 [NP,D]
__global__ void k_embed_pairs_f32(const float* __restrict__ tab, const int* __restrict__ sp,
                                  const int* __restrict__ nbr, float* __restrict__ dst, int D) {
  size_t i = (size_t)blockIdx.x * blockDim.x + threadIdx.x;
  if (i >= (size_t)NP * D) return;
  size_t p = i / D; int d = i % D;
  dst[i] = tab[(size_t)sp[nbr[p]] * D + d];
}
// table[species[a]] -> f32 [NA,D]
__global__ void k_embed_atoms_f32(const float* __restrict__ tab, const int* __restrict__ sp,
                                  float* __restrict__ dst, int D) {
  size_t i = (size_t)blockIdx.x * blockDim.x + threadIdx.x;
  if (i >= (size_t)NA * D) return;
  size_t a = i / D; int d = i % D;
  dst[i] = tab[(size_t)sp[a] * D + d];
}
// table[species[nbr[p]]] -> f16 concat block at column `off` of [NP,KIN]
__global__ void k_embed_pairs_h(const float* __restrict__ tab, const int* __restrict__ sp,
                                const int* __restrict__ nbr, _Float16* __restrict__ dst,
                                int KIN, int off) {
  size_t i = (size_t)blockIdx.x * blockDim.x + threadIdx.x;
  if (i >= (size_t)NP * 128) return;
  size_t p = i >> 7; int d = i & 127;
  dst[p * KIN + off + d] = (_Float16)tab[(size_t)sp[nbr[p]] * 128 + d];
}
// f32 [NP,128] -> f16 concat block
__global__ void k_copy_block_h(const float* __restrict__ src, _Float16* __restrict__ dst,
                               int KIN, int off) {
  size_t i = (size_t)blockIdx.x * blockDim.x + threadIdx.x;
  if (i >= (size_t)NP * 128) return;
  size_t p = i >> 7; int d = i & 127;
  dst[p * KIN + off + d] = (_Float16)src[i];
}
// geom(4) @ W(4,128) + b -> f16 concat block 0 (K=4 too small for WMMA)
__global__ void k_edge_embed_h(const float* __restrict__ geom, const float* __restrict__ W,
                               const float* __restrict__ b, _Float16* __restrict__ dst, int KIN) {
  size_t i = (size_t)blockIdx.x * blockDim.x + threadIdx.x;
  if (i >= (size_t)NP * 128) return;
  size_t p = i >> 7; int d = i & 127;
  const float* g = geom + p * 4;
  float acc = b[d] + g[0]*W[d] + g[1]*W[128+d] + g[2]*W[256+d] + g[3]*W[384+d];
  dst[p * KIN + d] = (_Float16)acc;
}

// ---------------- elementwise ----------------
__global__ void k_f2h(const float* __restrict__ x, _Float16* __restrict__ y, size_t n) {
  size_t i = (size_t)blockIdx.x * blockDim.x + threadIdx.x;
  if (i < n) y[i] = (_Float16)x[i];
}
__global__ void k_addf(float* __restrict__ dst, const float* __restrict__ src, size_t n) {
  size_t i = (size_t)blockIdx.x * blockDim.x + threadIdx.x;
  if (i < n) dst[i] += src[i];
}
__global__ void k_swiglu(const _Float16* __restrict__ h, _Float16* __restrict__ y,
                         int F, size_t rows) {
  size_t i = (size_t)blockIdx.x * blockDim.x + threadIdx.x;
  if (i >= rows * (size_t)F) return;
  size_t r = i / F; int f = i % F;
  float h1 = (float)h[r * 2 * F + f], h2 = (float)h[r * 2 * F + F + f];
  y[i] = (_Float16)(h1 * fsigmoid(h2));
}

// rmsnorm, one wave per row, f16 out
__global__ void k_rms(const float* __restrict__ x, const float* __restrict__ g,
                      _Float16* __restrict__ y, int D, int rows) {
  int row = blockIdx.x * (blockDim.x >> 5) + (threadIdx.x >> 5);
  int lane = threadIdx.x & 31;
  if (row >= rows) return;
  const float* xr = x + (size_t)row * D;
  float s = 0.f;
  for (int c = lane; c < D; c += 32) { float v = xr[c]; s += v * v; }
  #pragma unroll
  for (int m = 16; m; m >>= 1) s += __shfl_xor(s, m, 32);
  float rinv = rsqrtf(s / (float)D + 1e-6f);
  _Float16* yr = y + (size_t)row * D;
  for (int c = lane; c < D; c += 32) yr[c] = (_Float16)(xr[c] * rinv * g[c]);
}

// rmsnorm over token matrix built virtually from [proj | edge]
__global__ void k_rms_tokens(const float* __restrict__ proj, const float* __restrict__ edge,
                             const float* __restrict__ g, _Float16* __restrict__ y) {
  int row = blockIdx.x * (blockDim.x >> 5) + (threadIdx.x >> 5);
  int lane = threadIdx.x & 31;
  if (row >= MT) return;
  int a = row / TTOK, t = row % TTOK;
  const float* xr = (t == 0) ? proj + (size_t)a * 128
                             : edge + ((size_t)a * NNBR + t - 1) * 128;
  float s = 0.f;
  #pragma unroll
  for (int c = lane; c < 128; c += 32) { float v = xr[c]; s += v * v; }
  #pragma unroll
  for (int m = 16; m; m >>= 1) s += __shfl_xor(s, m, 32);
  float rinv = rsqrtf(s * (1.f / 128.f) + 1e-6f);
  _Float16* yr = y + (size_t)row * 128;
  #pragma unroll
  for (int c = lane; c < 128; c += 32) yr[c] = (_Float16)(xr[c] * rinv * g[c]);
}

// layernorm over concat([edge_msg, edge], -1), D=256, f16 out
__global__ void k_ln_concat(const float* __restrict__ a, const float* __restrict__ b,
                            const float* __restrict__ g, const float* __restrict__ bb,
                            _Float16* __restrict__ y) {
  int row = blockIdx.x * (blockDim.x >> 5) + (threadIdx.x >> 5);
  int lane = threadIdx.x & 31;
  if (row >= NP) return;
  const float* pa = a + (size_t)row * 128;
  const float* pb = b + (size_t)row * 128;
  float s1 = 0.f, s2 = 0.f;
  #pragma unroll
  for (int c = lane; c < 256; c += 32) {
    float v = (c < 128) ? pa[c] : pb[c - 128];
    s1 += v; s2 += v * v;
  }
  #pragma unroll
  for (int m = 16; m; m >>= 1) { s1 += __shfl_xor(s1, m, 32); s2 += __shfl_xor(s2, m, 32); }
  float mean = s1 * (1.f / 256.f);
  float var  = s2 * (1.f / 256.f) - mean * mean;
  float rinv = rsqrtf(var + 1e-5f);
  _Float16* yr = y + (size_t)row * 256;
  #pragma unroll
  for (int c = lane; c < 256; c += 32) {
    float v = (c < 128) ? pa[c] : pb[c - 128];
    yr[c] = (_Float16)((v - mean) * rinv * g[c] + bb[c]);
  }
}

// ---------------- attention (T=33, d=16, two-pass softmax, thread per (atom,head,q)) ------
// All q/k/v row addresses are 16B-aligned (token stride 768B, head stride 32B) -> b128 loads.
__global__ void k_attn(const _Float16* __restrict__ qkv, const float* __restrict__ logcut,
                       _Float16* __restrict__ out) {
  size_t i = (size_t)blockIdx.x * blockDim.x + threadIdx.x;
  if (i >= (size_t)NA * NHEADS * TTOK) return;
  int q = i % TTOK; int h = (i / TTOK) % NHEADS; int a = i / (TTOK * NHEADS);
  const _Float16* base = qkv + (size_t)a * TTOK * 384;
  const _Float16* qp = base + (size_t)q * 384 + h * 16;
  const float* lc = logcut + a * TTOK;
  V16H qv; qv.h[0] = *(const v8h*)qp; qv.h[1] = *(const v8h*)(qp + 8);
  float qr[16];
  #pragma unroll
  for (int d = 0; d < 16; ++d) qr[d] = (float)qv.v[d] * 0.25f;   // 1/sqrt(16)
  float mx = -1e30f;
  for (int k = 0; k < TTOK; ++k) {
    const _Float16* kp = base + (size_t)k * 384 + 128 + h * 16;
    V16H kv; kv.h[0] = *(const v8h*)kp; kv.h[1] = *(const v8h*)(kp + 8);
    float s = lc[k];
    #pragma unroll
    for (int d = 0; d < 16; ++d) s += qr[d] * (float)kv.v[d];
    mx = fmaxf(mx, s);
  }
  float Z = 0.f, o[16];
  #pragma unroll
  for (int d = 0; d < 16; ++d) o[d] = 0.f;
  for (int k = 0; k < TTOK; ++k) {
    const _Float16* kp = base + (size_t)k * 384 + 128 + h * 16;
    const _Float16* vp = base + (size_t)k * 384 + 256 + h * 16;
    V16H kv; kv.h[0] = *(const v8h*)kp; kv.h[1] = *(const v8h*)(kp + 8);
    V16H vv; vv.h[0] = *(const v8h*)vp; vv.h[1] = *(const v8h*)(vp + 8);
    float s = lc[k];
    #pragma unroll
    for (int d = 0; d < 16; ++d) s += qr[d] * (float)kv.v[d];
    float e = __expf(s - mx);
    Z += e;
    #pragma unroll
    for (int d = 0; d < 16; ++d) o[d] += e * (float)vv.v[d];
  }
  _Float16* orow = out + ((size_t)a * TTOK + q) * 128 + h * 16;
  float zi = 1.f / Z;
  #pragma unroll
  for (int d = 0; d < 16; ++d) orow[d] = (_Float16)(o[d] * zi);
}

// edge += attn[:,1:] ; tok0 gather to f16
__global__ void k_edge_attn_add(float* __restrict__ edge, const float* __restrict__ attn) {
  size_t i = (size_t)blockIdx.x * blockDim.x + threadIdx.x;
  if (i >= (size_t)NP * 128) return;
  size_t p = i >> 7; int d = i & 127;
  size_t a = p >> 5, j = p & 31;
  edge[i] += attn[((a * TTOK) + 1 + j) * 128 + d];
}
__global__ void k_tok0_h(const float* __restrict__ attn, _Float16* __restrict__ y) {
  size_t i = (size_t)blockIdx.x * blockDim.x + threadIdx.x;
  if (i >= (size_t)NA * 128) return;
  size_t a = i >> 7; int d = i & 127;
  y[i] = (_Float16)attn[(a * TTOK) * 128 + d];
}

// edge_msg[p] = msg[reverse[p]]
__global__ void k_permute(const float* __restrict__ msg, const int* __restrict__ rev,
                          float* __restrict__ out) {
  size_t i = (size_t)blockIdx.x * blockDim.x + threadIdx.x;
  if (i >= (size_t)NP * 128) return;
  size_t p = i >> 7; int d = i & 127;
  out[i] = msg[(size_t)rev[p] * 128 + d];
}

// row . w + b  (D=128, one wave per row)
__global__ void k_rowdot(const _Float16* __restrict__ x, const float* __restrict__ w,
                         const float* __restrict__ b, float* __restrict__ out, int rows) {
  int row = blockIdx.x * (blockDim.x >> 5) + (threadIdx.x >> 5);
  int lane = threadIdx.x & 31;
  if (row >= rows) return;
  float s = 0.f;
  #pragma unroll
  for (int c = lane; c < 128; c += 32) s += (float)x[(size_t)row * 128 + c] * w[c];
  #pragma unroll
  for (int m = 16; m; m >>= 1) s += __shfl_xor(s, m, 32);
  if (lane == 0) out[row] = s + b[0];
}

__global__ void k_final(const float* __restrict__ npred, const float* __restrict__ epred,
                        const float* __restrict__ cut, float* __restrict__ out) {
  int a = blockIdx.x * blockDim.x + threadIdx.x;
  if (a >= NA) return;
  float s = npred[a];
  #pragma unroll
  for (int j = 0; j < NNBR; ++j) s += epred[a * NNBR + j] * cut[a * NNBR + j];
  out[a] = s;
}

// =====================================================================
// Host orchestration
// =====================================================================
struct Lin { const float* w; const float* b; };
struct TransP { Lin cc, ce, cmi, cmo, mlp_in, mlp_out; const float *norm_attn,*norm_center,*norm_mlp; Lin out, qkv; };
struct LayerP { Lin comb1, comb2; const float *comb_nb,*comb_ng; Lin compress1, compress2, edge_embed; const float* neighbor_embed; TransP tr; };

extern "C" void kernel_launch(void* const* d_in, const int* in_sizes, int n_in,
                              void* d_out, int out_size, void* d_ws, size_t ws_size,
                              hipStream_t stream) {
  (void)in_sizes; (void)n_in; (void)out_size; (void)ws_size;
  const float* R_ij            = (const float*)d_in[0];
  const int* neighbors         = (const int*)d_in[2];
  const int* species           = (const int*)d_in[3];
  const int* reverse           = (const int*)d_in[4];
  const unsigned char* pairm   = (const unsigned char*)d_in[5];

  // params flattened in JAX pytree order (sorted dict keys; lin -> b then w)
  int idx = 7;
  auto F   = [&](int& i)->const float* { return (const float*)d_in[i++]; };
  auto LIN = [&](int& i)->Lin { Lin l; l.b = (const float*)d_in[i++]; l.w = (const float*)d_in[i++]; return l; };
  const float* edge_embedder = F(idx);
  LayerP L[2];
  for (int li = 0; li < 2; ++li) {
    L[li].comb1 = LIN(idx); L[li].comb2 = LIN(idx);
    L[li].comb_nb = F(idx); L[li].comb_ng = F(idx);
    L[li].compress1 = LIN(idx); L[li].compress2 = LIN(idx);
    L[li].edge_embed = LIN(idx);
    L[li].neighbor_embed = F(idx);
    TransP& t = L[li].tr;
    t.cc = LIN(idx); t.ce = LIN(idx); t.cmi = LIN(idx); t.cmo = LIN(idx);
    t.mlp_in = LIN(idx); t.mlp_out = LIN(idx);
    t.norm_attn = F(idx); t.norm_center = F(idx); t.norm_mlp = F(idx);
    t.out = LIN(idx); t.qkv = LIN(idx);
  }
  const float* node_embedder = F(idx);
  Lin edge_h1 = LIN(idx), edge_h2 = LIN(idx), edge_last = LIN(idx);
  Lin node_h1 = LIN(idx), node_h2 = LIN(idx), node_last = LIN(idx);

  // ---------------- workspace layout (bump) ----------------
  char* base = (char*)d_ws; size_t off = 0;
  auto B = [&](size_t bytes)->void* { void* p = base + off; off += (bytes + 255) & ~(size_t)255; return p; };
  float*     geomf    = (float*)B((size_t)NP * 4 * 4);
  float*     cutb     = (float*)B((size_t)NP * 4);
  float*     logcut   = (float*)B((size_t)NA * TTOK * 4);
  float*     edge_msg = (float*)B((size_t)NP * 128 * 4);
  float*     node_emb = (float*)B((size_t)NA * 512 * 4);
  float*     nodef    = (float*)B((size_t)NA * 512 * 4);
  float*     edgef    = (float*)B((size_t)NP * 128 * 4);
  float*     projf    = (float*)B((size_t)NA * 128 * 4);
  float*     ntmp     = (float*)B((size_t)NA * 512 * 4);
  float*     scrf     = (float*)B((size_t)MT * 128 * 4);       // 66 MB f32 scratch
  float*     npred    = (float*)B((size_t)NA * 4);
  float*     epred    = (float*)B((size_t)NP * 4);
  _Float16*  wt16     = (_Float16*)B((size_t)4 << 20);         // largest weight: 512x2048 f16 = 2MB
  _Float16*  big0     = (_Float16*)B((size_t)NP * 512 * 2);    // 128 MB f16 scratch A
  _Float16*  big1     = (_Float16*)B((size_t)NP * 256 * 2);    // 64 MB f16 scratch B
  float*     pred     = (float*)d_out;

  auto EW = [&](size_t n) { return dim3((unsigned)CDIV(n, (size_t)256)); };    // elementwise grid, 256 thr
  auto RW = [&](size_t rows) { return dim3((unsigned)CDIV(rows, (size_t)4)); };// wave-per-row grid, 128 thr

  // convert weight + run WMMA gemm (64x64 register-blocked when shape allows)
  auto GEMM = [&](const _Float16* A, Lin lin, int K, int N, int M, int act,
                  float* Cf, _Float16* Ch) {
    k_wt_conv<<<CDIV(K * N, 256), 256, 0, stream>>>(lin.w, wt16, K, N);
    dim3 b(128);
    if ((M & 63) == 0 && (N & 63) == 0) {
      int waves = (M >> 6) * (N >> 6);
      dim3 g(CDIV(waves, 4));
      if (act) k_gemm64<1><<<g, b, 0, stream>>>(A, wt16, lin.b, Cf, Ch, M, N, K);
      else     k_gemm64<0><<<g, b, 0, stream>>>(A, wt16, lin.b, Cf, Ch, M, N, K);
    } else {
      int waves = (M >> 4) * (N >> 4);
      dim3 g(CDIV(waves, 4));
      if (act) k_gemm16<1><<<g, b, 0, stream>>>(A, wt16, lin.b, Cf, Ch, M, N, K);
      else     k_gemm16<0><<<g, b, 0, stream>>>(A, wt16, lin.b, Cf, Ch, M, N, K);
    }
  };

  // ---------------- prep ----------------
  k_pair_prep<<<CDIV(NP, 256), 256, 0, stream>>>(R_ij, pairm, geomf, cutb);
  k_logcut<<<CDIV(NA * TTOK, 256), 256, 0, stream>>>(cutb, logcut);
  k_embed_pairs_f32<<<EW((size_t)NP * 128), 256, 0, stream>>>(edge_embedder, species, neighbors, edge_msg, 128);
  k_embed_atoms_f32<<<EW((size_t)NA * 512), 256, 0, stream>>>(node_embedder, species, node_emb, 512);

  // ---------------- GNN layers ----------------
  for (int li = 0; li < 2; ++li) {
    const LayerP& lp = L[li];
    const TransP& t  = lp.tr;
    const int KIN = (li == 0) ? 256 : 384;

    // build concat input [g | (neighbor_embed) | edge_msg] as f16 in big0
    k_edge_embed_h<<<EW((size_t)NP * 128), 256, 0, stream>>>(geomf, lp.edge_embed.w, lp.edge_embed.b, big0, KIN);
    int coff = 128;
    if (li == 1) {
      k_embed_pairs_h<<<EW((size_t)NP * 128), 256, 0, stream>>>(lp.neighbor_embed, species, neighbors, big0, KIN, coff);
      coff += 128;
    }
    k_copy_block_h<<<EW((size_t)NP * 128), 256, 0, stream>>>(edge_msg, big0, KIN, coff);

    // edge = compress2(silu(compress1(inp)))
    GEMM(big0, lp.compress1, KIN, 128, NP, 1, nullptr, big1);
    GEMM(big1, lp.compress2, 128, 128, NP, 0, edgef, nullptr);

    // node = node_emb
    hipMemcpyAsync(nodef, node_emb, (size_t)NA * 512 * 4, hipMemcpyDeviceToDevice, stream);

    // ---- transformer ----
    // proj = center_contract(node)
    k_f2h<<<EW((size_t)NA * 512), 256, 0, stream>>>(nodef, big1, (size_t)NA * 512);
    GEMM(big1, t.cc, 512, 128, NA, 0, projf, nullptr);
    // tokens = rmsnorm(concat([proj, edge])) -> f16 big1 [MT,128]
    k_rms_tokens<<<RW(MT), 128, 0, stream>>>(projf, edgef, t.norm_attn, big1);
    // qkv -> f16 big0 [MT,384]
    GEMM(big1, t.qkv, 128, 384, MT, 0, nullptr, big0);
    // attention -> f16 big1 [MT,128]
    k_attn<<<EW((size_t)NA * NHEADS * TTOK), 256, 0, stream>>>(big0, logcut, big1);
    // out projection -> f32 scrf [MT,128]
    GEMM(big1, t.out, 128, 128, MT, 0, scrf, nullptr);
    // out_edge = edge + attn[:,1:]
    k_edge_attn_add<<<EW((size_t)NP * 128), 256, 0, stream>>>(edgef, scrf);
    // out_node = node + center_expand(attn[:,0])
    k_tok0_h<<<EW((size_t)NA * 128), 256, 0, stream>>>(scrf, big0);
    GEMM(big0, t.ce, 128, 512, NA, 0, ntmp, nullptr);
    k_addf<<<EW((size_t)NA * 512), 256, 0, stream>>>(nodef, ntmp, (size_t)NA * 512);
    // center swiglu MLP: 512 -> 2048 -> gate 1024 -> 512
    k_rms<<<RW(NA), 128, 0, stream>>>(nodef, t.norm_center, big1, 512, NA);
    GEMM(big1, t.cmi, 512, 2048, NA, 0, nullptr, big0);
    k_swiglu<<<EW((size_t)NA * 1024), 256, 0, stream>>>(big0, big1, 1024, NA);
    GEMM(big1, t.cmo, 1024, 512, NA, 0, ntmp, nullptr);
    k_addf<<<EW((size_t)NA * 512), 256, 0, stream>>>(nodef, ntmp, (size_t)NA * 512);
    // edge swiglu MLP: 128 -> 512 -> gate 256 -> 128
    k_rms<<<RW(NP), 128, 0, stream>>>(edgef, t.norm_mlp, big1, 128, NP);
    GEMM(big1, t.mlp_in, 128, 512, NP, 0, nullptr, big0);
    k_swiglu<<<EW((size_t)NP * 256), 256, 0, stream>>>(big0, big1, 256, NP);
    GEMM(big1, t.mlp_out, 256, 128, NP, 0, scrf, nullptr);
    k_addf<<<EW((size_t)NP * 128), 256, 0, stream>>>(edgef, scrf, (size_t)NP * 128);

    if (li == 0) {
      // message combine + reverse-permute scatter
      k_ln_concat<<<RW(NP), 128, 0, stream>>>(edge_msg, edgef, lp.comb_ng, lp.comb_nb, big1);
      GEMM(big1, lp.comb1, 256, 256, NP, 1, nullptr, big0);
      GEMM(big0, lp.comb2, 256, 128, NP, 0, scrf, nullptr);
      k_permute<<<EW((size_t)NP * 128), 256, 0, stream>>>(scrf, reverse, edge_msg);
    } else {
      // readout: node branch
      k_f2h<<<EW((size_t)NA * 512), 256, 0, stream>>>(nodef, big1, (size_t)NA * 512);
      GEMM(big1, node_h1, 512, 128, NA, 1, nullptr, big0);
      GEMM(big0, node_h2, 128, 128, NA, 1, nullptr, big1);
      k_rowdot<<<RW(NA), 128, 0, stream>>>(big1, node_last.w, node_last.b, npred, NA);
      // readout: edge branch
      k_f2h<<<EW((size_t)NP * 128), 256, 0, stream>>>(edgef, big0, (size_t)NP * 128);
      GEMM(big0, edge_h1, 128, 128, NP, 1, nullptr, big1);
      GEMM(big1, edge_h2, 128, 128, NP, 1, nullptr, big0);
      k_rowdot<<<RW(NP), 128, 0, stream>>>(big0, edge_last.w, edge_last.b, epred, NP);
      k_final<<<CDIV(NA, 256), 256, 0, stream>>>(npred, epred, cutb, pred);
    }
  }
}
